// QuantLinearLUT_24000277250518
// MI455X (gfx1250) — matrix operand order repaired and checked by reading, not compiled, not run on hardware
//
#include <hip/hip_runtime.h>

// QuantLinearLUT as bf16 WMMA GEMM for gfx1250 (MI455X).
// M=4096, K=4096, N=11008 -> compute-bound (AI ~1370 FLOP/B).
// v2: BM=256 (halves redundant weight dequant), byte->bf16x2 pair LUT in LDS
// (halves dequant LDS ops, removes in-loop cvts), explicit double-buffered
// software pipeline (global loads overlap WMMA).

typedef __attribute__((ext_vector_type(16))) __bf16 v16bf;
typedef __attribute__((ext_vector_type(8)))  __bf16 v8bf;
typedef __attribute__((ext_vector_type(4)))  __bf16 v4bf;
typedef __attribute__((ext_vector_type(2)))  __bf16 v2bf;
typedef __attribute__((ext_vector_type(8)))  float  v8f;

#define K_DIM 4096
#define N_DIM 11008
#define BM 256
#define BN 128
#define BK 64
#define LDSS 72            // BK + 8 bf16 pad (16B) for bank spread
#define THREADS 512        // 16 waves: 8 along M x 2 along N
#define KSTEPS (K_DIM / BK)

__global__ __launch_bounds__(THREADS)
void qlut_gemm_bf16_wmma(const float* __restrict__ x,
                         const int*   __restrict__ qw,
                         const float* __restrict__ lut,
                         const float* __restrict__ bias,
                         float*       __restrict__ out)
{
    __shared__ __bf16 As[2][BM][LDSS];        // A tiles  [m][k] bf16, double buffered
    __shared__ __bf16 Bs[2][BN][LDSS];        // W^T tiles [n][k] bf16, double buffered
    __shared__ unsigned int Lut2[BN][256];    // byte -> packed bf16 pair (k even | k odd << 16)
    __shared__ float  Lutf[BN][16];           // raw per-channel LUT (coalesced staging)

    const int tid  = threadIdx.x;
    const int lane = tid & 31;
    const int wid  = tid >> 5;
    const int l16  = lane & 15;
    const int hi   = lane >> 4;

    const int n0 = blockIdx.x * BN;
    const int m0 = blockIdx.y * BM;

    const int wm = wid & 7;                   // 8 M-slices of 32 rows
    const int wn = wid >> 3;                  // 2 N-slices of 64 cols

    // per-thread staging coords
    const int ac4 = (tid & 15) * 4;           // A col (float4 granularity)
    const int pr  = tid >> 6;                 // qweight k-row within step (0..7)
    const int nq  = (tid & 63) * 2;           // 2 consecutive output channels

    // ---- stage raw LUTs coalesced, then build byte->bf16x2 pair table ----
    #pragma unroll
    for (int i = 0; i < 4; ++i) {
        int idx = tid + i * THREADS;          // 0..2047
        Lutf[idx >> 4][idx & 15] = lut[(size_t)(n0 + (idx >> 4)) * 16 + (idx & 15)];
    }
    __syncthreads();
    #pragma unroll
    for (int i = 0; i < 64; ++i) {
        int idx = tid + i * THREADS;          // 0..32767
        int ch = idx >> 8, b = idx & 255;
        v2bf p;
        p[0] = (__bf16)Lutf[ch][b & 15];      // low nibble -> even K
        p[1] = (__bf16)Lutf[ch][b >> 4];      // high nibble -> odd K
        Lut2[ch][b] = __builtin_bit_cast(unsigned int, p);
    }

    v8f acc[2][4];
    #pragma unroll
    for (int mt = 0; mt < 2; ++mt)
        #pragma unroll
        for (int nt = 0; nt < 4; ++nt)
            acc[mt][nt] = (v8f)0.0f;

    float4 a_reg[8];
    int2   w_reg;

    // ---- prologue: load k-step 0 into registers ----
    #pragma unroll
    for (int i = 0; i < 8; ++i) {
        int idx = tid + i * THREADS;          // 0..4095
        int r = idx >> 4;                     // 0..255
        a_reg[i] = *(const float4*)&x[(size_t)(m0 + r) * K_DIM + ac4];
    }
    w_reg = *(const int2*)&qw[(size_t)pr * N_DIM + n0 + nq];

    __syncthreads();                          // Lut2 complete

    // ---- store k-step 0 into buffer 0 ----
    #pragma unroll
    for (int i = 0; i < 8; ++i) {
        int idx = tid + i * THREADS;
        int r = idx >> 4;
        v4bf bb;
        bb[0] = (__bf16)a_reg[i].x; bb[1] = (__bf16)a_reg[i].y;
        bb[2] = (__bf16)a_reg[i].z; bb[3] = (__bf16)a_reg[i].w;
        *(v4bf*)&As[0][r][ac4] = bb;
    }
    {
        const unsigned int w01[2] = { (unsigned int)w_reg.x, (unsigned int)w_reg.y };
        #pragma unroll
        for (int j = 0; j < 2; ++j) {
            const unsigned int u = w01[j];
            const int n = nq + j;
            uint4 d;
            d.x = Lut2[n][u & 255];
            d.y = Lut2[n][(u >> 8) & 255];
            d.z = Lut2[n][(u >> 16) & 255];
            d.w = Lut2[n][u >> 24];
            *(uint4*)&Bs[0][n][pr * 8] = d;   // 8 bf16 along K, 16B aligned
        }
    }
    __syncthreads();

    // ---- main pipelined loop ----
    for (int kb = 0; kb < KSTEPS; ++kb) {
        const int cur = kb & 1;
        const bool more = (kb + 1) < KSTEPS;

        // (1) issue next k-step global loads into registers (overlap with WMMA)
        if (more) {
            const int k1 = (kb + 1) * BK;
            #pragma unroll
            for (int i = 0; i < 8; ++i) {
                int idx = tid + i * THREADS;
                int r = idx >> 4;
                a_reg[i] = *(const float4*)&x[(size_t)(m0 + r) * K_DIM + k1 + ac4];
            }
            w_reg = *(const int2*)&qw[(size_t)(k1 / 8 + pr) * N_DIM + n0 + nq];
        }
        if (kb + 2 < KSTEPS) {
            __builtin_prefetch(&x[(size_t)(m0 + (tid >> 4)) * K_DIM + (kb + 2) * BK + ac4], 0, 0);
            __builtin_prefetch(&qw[(size_t)((kb + 2) * (BK / 8) + pr) * N_DIM + n0 + nq], 0, 0);
        }

        // (2) compute on current buffer: 2 K-chunks of 32, 2x4 WMMA tiles per wave
        #pragma unroll
        for (int kk = 0; kk < BK; kk += 32) {
            v16bf af[2];
            #pragma unroll
            for (int mt = 0; mt < 2; ++mt) {
                // A layout (16-bit 16x32): lanes 0-15 K 0..7 & 16..23,
                // lanes 16-31 K 8..15 & 24..31, M = lane%16.
                const int row = wm * 32 + mt * 16 + l16;
                const int col = kk + hi * 8;
                v8bf lo = *(const v8bf*)&As[cur][row][col];
                v8bf hh = *(const v8bf*)&As[cur][row][col + 16];
                af[mt] = __builtin_shufflevector(lo, hh,
                         0,1,2,3,4,5,6,7,8,9,10,11,12,13,14,15);
            }
            v16bf bfr[4];
            #pragma unroll
            for (int nt = 0; nt < 4; ++nt) {
                // B layout (16-bit 32x16): N = lane%16, lanes 0-15 K 0..7 & 16..23,
                // lanes 16-31 K 8..15 & 24..31 (matches A).
                const int ch = wn * 64 + nt * 16 + l16;
                const int kc = kk + hi * 8;
                v8bf lo = *(const v8bf*)&Bs[cur][ch][kc];
                v8bf hh = *(const v8bf*)&Bs[cur][ch][kc + 16];
                bfr[nt] = __builtin_shufflevector(lo, hh,
                          0,1,2,3,4,5,6,7,8,9,10,11,12,13,14,15);
            }
            #pragma unroll
            for (int mt = 0; mt < 2; ++mt)
                #pragma unroll
                for (int nt = 0; nt < 4; ++nt)
                    acc[mt][nt] = __builtin_amdgcn_wmma_f32_16x16x32_bf16(
                        false, af[mt], false, bfr[nt],
                        (short)0, acc[mt][nt], false, false);
        }

        // (3) convert/dequant staged registers into the other buffer
        if (more) {
            const int nxt = cur ^ 1;
            #pragma unroll
            for (int i = 0; i < 8; ++i) {
                int idx = tid + i * THREADS;
                int r = idx >> 4;
                v4bf bb;
                bb[0] = (__bf16)a_reg[i].x; bb[1] = (__bf16)a_reg[i].y;
                bb[2] = (__bf16)a_reg[i].z; bb[3] = (__bf16)a_reg[i].w;
                *(v4bf*)&As[nxt][r][ac4] = bb;
            }
            const unsigned int w01[2] = { (unsigned int)w_reg.x, (unsigned int)w_reg.y };
            #pragma unroll
            for (int j = 0; j < 2; ++j) {
                const unsigned int u = w01[j];
                const int n = nq + j;
                uint4 d;
                d.x = Lut2[n][u & 255];
                d.y = Lut2[n][(u >> 8) & 255];
                d.z = Lut2[n][(u >> 16) & 255];
                d.w = Lut2[n][u >> 24];
                *(uint4*)&Bs[nxt][n][pr * 8] = d;
            }
        }
        __syncthreads();
    }

    // ---- epilogue: bias add + store (C layout: VGPR v -> M = v + 8*hi, N = lane%16) ----
    const int mrow0 = m0 + wm * 32;
    const int ncol0 = n0 + wn * 64;
    #pragma unroll
    for (int nt = 0; nt < 4; ++nt) {
        const int n  = ncol0 + nt * 16 + l16;
        const float bn = bias[n];
        #pragma unroll
        for (int mt = 0; mt < 2; ++mt) {
            const int mbase = mrow0 + mt * 16 + hi * 8;
            const v8f a = acc[mt][nt];
            #pragma unroll
            for (int v = 0; v < 8; ++v)
                out[(size_t)(mbase + v) * N_DIM + n] = a[v] + bn;
        }
    }
}

extern "C" void kernel_launch(void* const* d_in, const int* in_sizes, int n_in,
                              void* d_out, int out_size, void* d_ws, size_t ws_size,
                              hipStream_t stream) {
    (void)n_in; (void)out_size; (void)d_ws; (void)ws_size;
    const float* x    = (const float*)d_in[0];
    const int*   qw   = (const int*)  d_in[1];
    const float* lut  = (const float*)d_in[2];
    const float* bias = (const float*)d_in[3];
    float*       out  = (float*)d_out;

    const int M = in_sizes[0] / K_DIM;        // 4096
    dim3 grid(N_DIM / BN, M / BM);            // 86 x 16
    dim3 block(THREADS);
    qlut_gemm_bf16_wmma<<<grid, block, 0, stream>>>(x, qw, lut, bias, out);
}
